// ResContextBlock_49392123904122
// MI455X (gfx1250) — matrix operand order repaired and checked by reading, not compiled, not run on hardware
//
#include <hip/hip_runtime.h>
#include <hip/hip_bf16.h>

typedef __attribute__((ext_vector_type(16))) _Float16 v16h;
typedef __attribute__((ext_vector_type(8)))  _Float16 v8h;
typedef __attribute__((ext_vector_type(8)))  float    v8f;

#define N_VOX 200000
#define CH 64
#define KOFF 9
// per-weight fragment storage: 9 offsets * 2 K-chunks * 4 N-tiles * 512 halves
#define FRAG_HALF_PER_W (9 * 2 * 4 * 512)   // 36864 halves = 73728 B
#define EPS_BN 1e-5f
#define LSLOPE 0.01f

// ---------------------------------------------------------------------------
// Convert fp32 weights [9][Cin=64][Cout=64] into per-lane WMMA B fragments for
// v_wmma_f32_16x16x32_f16. ISA B layout (32x16 f16): lane L holds column
// n = L&15; packed half h=0..15 holds K = (h>>3)*16 + (L>>4)*8 + (h&7) of the 32-wide chunk.
// LDS-friendly storage (conflict-free ds_load_b128, 16B lane stride):
//   half_off(fid,lane,h) = fid*512 + (h>>3)*256 + lane*8 + (h&7)
// fid = (koff*2 + kk)*4 + t.
// ---------------------------------------------------------------------------
__global__ void prep_weights_kernel(const float* __restrict__ w0,
                                    const float* __restrict__ w1,
                                    const float* __restrict__ w2,
                                    const float* __restrict__ w3,
                                    _Float16* __restrict__ wfrag) {
  int e = blockIdx.x * blockDim.x + threadIdx.x;
  if (e >= 4 * FRAG_HALF_PER_W) return;
  int wsel   = e / FRAG_HALF_PER_W;
  int rem    = e % FRAG_HALF_PER_W;
  int fid    = rem >> 9;               // 0..71
  int within = rem & 511;
  int part   = within >> 8;            // 0 = halves 0..7, 1 = halves 8..15
  int lane   = (within >> 3) & 31;
  int h      = part * 8 + (within & 7);
  int koff = fid >> 3;
  int kk   = (fid >> 2) & 1;
  int t    = fid & 3;
  int cin  = kk * 32 + (h >> 3) * 16 + ((lane >> 4) & 1) * 8 + (h & 7);
  int cout = t * 16 + (lane & 15);
  const float* src = (wsel == 0) ? w0 : (wsel == 1) ? w1 : (wsel == 2) ? w2 : w3;
  wfrag[e] = (_Float16)src[(koff * CH + cin) * CH + cout];
}

// fp32 features -> f16, padded buffer [N+1][64] with row N == 0 (gather pad row)
__global__ void cvt_features_kernel(const float* __restrict__ f,
                                    _Float16* __restrict__ out) {
  long i = (long)blockIdx.x * blockDim.x + threadIdx.x;
  if (i >= (long)(N_VOX + 1) * CH) return;
  int r = (int)(i >> 6);
  out[i] = (r < N_VOX) ? (_Float16)f[i] : (_Float16)0.0f;
}

// ---------------------------------------------------------------------------
// Gather + WMMA submanifold conv. One wave per 16-row M strip; 8 waves/block.
// Weights staged in LDS once per block (72KB; DS pipe decoupled from gathers).
// A fragment (16x32 f16): lane L = row L&15; v0..3 = K hi*8+0..7,
// v4..7 = K 16+hi*8+0..7  -> two contiguous 16B global loads per K chunk.
// ---------------------------------------------------------------------------
__global__ __launch_bounds__(256)
void subm_conv_kernel(const _Float16* __restrict__ src,    // [N+1][64], row N = 0
                      const int*      __restrict__ nbr,    // [N][9]
                      const _Float16* __restrict__ wfrag,  // FRAG_HALF_PER_W halves
                      _Float16*       __restrict__ zout) { // [>=N][64]
  __shared__ _Float16 lw[FRAG_HALF_PER_W];                 // 73728 B

  // cooperative weight stage: 4608 x b128, 18 per thread, no divergence
  {
    const v8h* gsrc = (const v8h*)wfrag;
    v8h* ldst = (v8h*)lw;
#pragma unroll
    for (int i = 0; i < FRAG_HALF_PER_W / 8 / 256; ++i)
      ldst[i * 256 + threadIdx.x] = gsrc[i * 256 + threadIdx.x];
  }
  __syncthreads();

  const int lane = threadIdx.x & 31;
  const int wave = threadIdx.x >> 5;
  const int base = blockIdx.x * 128 + wave * 16;
  const int mrow = base + (lane & 15);
  const int hi   = (lane >> 4) & 1;
  const bool rowok = (mrow < N_VOX);
  const int rr = rowok ? mrow : 0;

  v8f acc[4] = {};
  const _Float16* lwl = lw + lane * 8;     // lane-local LDS base (16B stride)

  for (int k = 0; k < KOFF; ++k) {
    int idx = nbr[rr * KOFF + k];
    idx = rowok ? idx : N_VOX;             // pad row for out-of-range strips
    const _Float16* rowp = src + (size_t)idx * CH;
    // hoist both K-chunk A fragments (4 x b128) to the loop head
    v8h a0lo = *(const v8h*)(rowp + hi * 8);
    v8h a0hi = *(const v8h*)(rowp + 16 + hi * 8);
    v8h a1lo = *(const v8h*)(rowp + 32 + hi * 8);
    v8h a1hi = *(const v8h*)(rowp + 48 + hi * 8);
#pragma unroll
    for (int kk = 0; kk < 2; ++kk) {
      v16h A;
#pragma unroll
      for (int i = 0; i < 8; ++i) {
        A[i]     = kk ? a1lo[i] : a0lo[i];
        A[i + 8] = kk ? a1hi[i] : a0hi[i];
      }
      const int fbase = (k * 2 + kk) * 4;
      // batch all 8 ds_load_b128 for the 4 B fragments before the WMMAs
      v8h blo[4], bhi[4];
#pragma unroll
      for (int t = 0; t < 4; ++t) {
        const _Float16* bp = lwl + (fbase + t) * 512;
        blo[t] = *(const v8h*)(bp);
        bhi[t] = *(const v8h*)(bp + 256);
      }
#pragma unroll
      for (int t = 0; t < 4; ++t) {
        v16h B;
#pragma unroll
        for (int i = 0; i < 8; ++i) { B[i] = blo[t][i]; B[i + 8] = bhi[t][i]; }
        acc[t] = __builtin_amdgcn_wmma_f32_16x16x32_f16(
            false, A, false, B, (short)0, acc[t], false, false);
      }
    }
  }

  // C/D layout: lane L, vgpr j -> row = j + hi*8, col = t*16 + (L&15)
#pragma unroll
  for (int j = 0; j < 8; ++j) {
    int row = base + j + hi * 8;
    if (row < N_VOX) {
#pragma unroll
      for (int t = 0; t < 4; ++t) {
        float v = acc[t][j];
        v = (v > 0.0f) ? v : LSLOPE * v;      // LeakyReLU before BN
        zout[(size_t)row * CH + t * 16 + (lane & 15)] = (_Float16)v;
      }
    }
  }
}

__global__ void zero_stats_kernel(float* p) { p[threadIdx.x] = 0.0f; }

// per-channel sum / sum-of-squares over N rows (f32 accumulation)
__global__ __launch_bounds__(256)
void stats_kernel(const _Float16* __restrict__ z, float* __restrict__ sums) {
  __shared__ float ls[256], lq[256];
  const int tid = threadIdx.x;
  const int c   = tid & 63;
  const int rg  = tid >> 6;                 // 4 rows processed per iter
  float s = 0.0f, q = 0.0f;
  for (int r = blockIdx.x * 4 + rg; r < N_VOX; r += gridDim.x * 4) {
    float v = (float)z[(size_t)r * CH + c];
    s += v; q += v * v;
  }
  ls[tid] = s; lq[tid] = q;
  __syncthreads();
  if (tid < 64) {
    s = ls[tid] + ls[tid + 64] + ls[tid + 128] + ls[tid + 192];
    q = lq[tid] + lq[tid + 64] + lq[tid + 128] + lq[tid + 192];
    atomicAdd(&sums[tid], s);
    atomicAdd(&sums[64 + tid], q);
  }
}

// fold gamma/beta/mean/var into per-channel scale+bias: y = z*sb[c] + sb[64+c]
__global__ void finalize_kernel(const float* __restrict__ sums,
                                const float* __restrict__ gamma,
                                const float* __restrict__ beta,
                                float* __restrict__ sb) {
  int c = threadIdx.x;                       // 64 threads
  const float invn = 1.0f / (float)N_VOX;
  float mean = sums[c] * invn;
  float var  = sums[64 + c] * invn - mean * mean;   // biased var, matches ref
  float s    = gamma[c] * rsqrtf(var + EPS_BN);
  sb[c]      = s;
  sb[64 + c] = beta[c] - mean * s;
}

// in-place BN affine on padded f16 buffer; forces pad row N to zero
__global__ void normalize_kernel(_Float16* __restrict__ z,
                                 const float* __restrict__ sb) {
  long i = (long)blockIdx.x * blockDim.x + threadIdx.x;
  if (i >= (long)(N_VOX + 1) * CH) return;
  int r = (int)(i >> 6);
  int c = (int)(i & 63);
  float v = (r < N_VOX) ? ((float)z[i] * sb[c] + sb[64 + c]) : 0.0f;
  z[i] = (_Float16)v;
}

// out = BN(z4) + BN(z2) in fp32
__global__ void final_add_kernel(const _Float16* __restrict__ z2,
                                 const float* __restrict__ sb2,
                                 const _Float16* __restrict__ z4,
                                 const float* __restrict__ sb4,
                                 float* __restrict__ out) {
  long i = (long)blockIdx.x * blockDim.x + threadIdx.x;
  if (i >= (long)N_VOX * CH) return;
  int c = (int)(i & 63);
  out[i] = ((float)z4[i] * sb4[c] + sb4[64 + c]) +
           ((float)z2[i] * sb2[c] + sb2[64 + c]);
}

extern "C" void kernel_launch(void* const* d_in, const int* in_sizes, int n_in,
                              void* d_out, int out_size, void* d_ws, size_t ws_size,
                              hipStream_t stream) {
  const float* features = (const float*)d_in[0];
  const float* w1    = (const float*)d_in[1];
  const float* w1_2  = (const float*)d_in[2];
  const float* w2    = (const float*)d_in[3];
  const float* w3    = (const float*)d_in[4];
  const float* g0    = (const float*)d_in[5];
  const float* b0    = (const float*)d_in[6];
  const float* g0_2  = (const float*)d_in[7];
  const float* b0_2  = (const float*)d_in[8];
  const float* g1    = (const float*)d_in[9];
  const float* b1    = (const float*)d_in[10];
  const float* g2    = (const float*)d_in[11];
  const float* b2    = (const float*)d_in[12];
  const int*   nbr13 = (const int*)d_in[13];
  const int*   nbr31 = (const int*)d_in[14];
  float* out = (float*)d_out;

  // ---- workspace carve-up (~103 MB) ----
  char* ws = (char*)d_ws;
  const size_t FHA = (((size_t)(N_VOX + 1) * CH * sizeof(_Float16)) + 255) & ~(size_t)255;
  size_t off = 0;
  _Float16* wfrag = (_Float16*)(ws + off);
  off += ((size_t)4 * FRAG_HALF_PER_W * sizeof(_Float16) + 255) & ~(size_t)255;
  _Float16* f0h = (_Float16*)(ws + off); off += FHA;   // converted input
  _Float16* zA  = (_Float16*)(ws + off); off += FHA;   // reused intermediate
  _Float16* z2  = (_Float16*)(ws + off); off += FHA;   // shortcut pre-BN
  _Float16* z4  = (_Float16*)(ws + off); off += FHA;   // main pre-BN
  float* sums = (float*)(ws + off); off += 512;        // 128 f32, reused
  float* sb1  = (float*)(ws + off); off += 512;
  float* sb2  = (float*)(ws + off); off += 512;
  float* sb3  = (float*)(ws + off); off += 512;
  float* sb4  = (float*)(ws + off); off += 512;
  (void)ws_size; (void)in_sizes; (void)n_in; (void)out_size;

  const int prepGrid = (4 * FRAG_HALF_PER_W + 255) / 256;
  const long padTot  = (long)(N_VOX + 1) * CH;
  const int  padGrid = (int)((padTot + 255) / 256);
  const long outTot  = (long)N_VOX * CH;
  const int  outGrid = (int)((outTot + 255) / 256);
  const int  convGrid = (N_VOX + 127) / 128;   // 128 rows / block (8 waves x 16)
  const int  STATG = 512;

  prep_weights_kernel<<<prepGrid, 256, 0, stream>>>(w1, w1_2, w2, w3, wfrag);
  cvt_features_kernel<<<padGrid, 256, 0, stream>>>(features, f0h);

  // shortcut: conv(nbr13,w1) -> BN(g0,b0) -> conv(nbr31,w1_2) -> BN(g0_2,b0_2)
  subm_conv_kernel<<<convGrid, 256, 0, stream>>>(f0h, nbr13, wfrag + 0 * FRAG_HALF_PER_W, zA);
  zero_stats_kernel<<<1, 128, 0, stream>>>(sums);
  stats_kernel<<<STATG, 256, 0, stream>>>(zA, sums);
  finalize_kernel<<<1, 64, 0, stream>>>(sums, g0, b0, sb1);
  normalize_kernel<<<padGrid, 256, 0, stream>>>(zA, sb1);

  subm_conv_kernel<<<convGrid, 256, 0, stream>>>(zA, nbr31, wfrag + 1 * FRAG_HALF_PER_W, z2);
  zero_stats_kernel<<<1, 128, 0, stream>>>(sums);
  stats_kernel<<<STATG, 256, 0, stream>>>(z2, sums);
  finalize_kernel<<<1, 64, 0, stream>>>(sums, g0_2, b0_2, sb2);   // affine fused at end

  // main: conv(nbr31,w2) -> BN(g1,b1) -> conv(nbr13,w3) -> BN(g2,b2)
  subm_conv_kernel<<<convGrid, 256, 0, stream>>>(f0h, nbr31, wfrag + 2 * FRAG_HALF_PER_W, zA);
  zero_stats_kernel<<<1, 128, 0, stream>>>(sums);
  stats_kernel<<<STATG, 256, 0, stream>>>(zA, sums);
  finalize_kernel<<<1, 64, 0, stream>>>(sums, g1, b1, sb3);
  normalize_kernel<<<padGrid, 256, 0, stream>>>(zA, sb3);

  subm_conv_kernel<<<convGrid, 256, 0, stream>>>(zA, nbr13, wfrag + 3 * FRAG_HALF_PER_W, z4);
  zero_stats_kernel<<<1, 128, 0, stream>>>(sums);
  stats_kernel<<<STATG, 256, 0, stream>>>(z4, sums);
  finalize_kernel<<<1, 64, 0, stream>>>(sums, g2, b2, sb4);

  final_add_kernel<<<outGrid, 256, 0, stream>>>(z2, sb2, z4, sb4, out);
}
